// GCN_60550448939055
// MI455X (gfx1250) — matrix operand order repaired and checked by reading, not compiled, not run on hardware
//
#include <hip/hip_runtime.h>
#include <math.h>

typedef __bf16 bf16_t;
typedef __attribute__((ext_vector_type(16))) __bf16 v16bf;
typedef __attribute__((ext_vector_type(8)))  __bf16 v8bf;
typedef __attribute__((ext_vector_type(8)))  float  v8f;

#define NNODES 100000
#define NEDGES 1600000
#define NQ     500000
#define DD     256
#define QCH    62500   // 8 chunks of queries

union AFrag { v16bf v; v8bf h[2]; };

// ---------------- degree / normalization ----------------
__global__ void k_deg_init(float* deg) {
    int i = blockIdx.x * blockDim.x + threadIdx.x;
    if (i < NNODES) deg[i] = 1.0f;   // self loop
}
__global__ void k_deg_acc(const int* __restrict__ row, float* deg) {
    int e = blockIdx.x * blockDim.x + threadIdx.x;
    if (e < NEDGES) atomicAdd(&deg[row[e]], 1.0f);
}
__global__ void k_dinv(float* d) {
    int i = blockIdx.x * blockDim.x + threadIdx.x;
    if (i < NNODES) d[i] = rsqrtf(d[i]);   // deg >= 1 always
}

// ---------------- conversions ----------------
__global__ void k_f32_to_bf16(const float* __restrict__ in, bf16_t* __restrict__ out, int n) {
    int i = blockIdx.x * blockDim.x + threadIdx.x;
    if (i < n) out[i] = (bf16_t)in[i];
}
__global__ void k_bias_relu_bf16(const float* __restrict__ agg, const float* __restrict__ b,
                                 bf16_t* __restrict__ out) {
    int i = blockIdx.x * blockDim.x + threadIdx.x;
    if (i < NNODES * DD) {
        float v = agg[i] + b[i & (DD - 1)];
        out[i] = (bf16_t)(v > 0.0f ? v : 0.0f);
    }
}
__global__ void k_bias_f32(const float* __restrict__ agg, const float* __restrict__ b,
                           float* __restrict__ out) {
    int i = blockIdx.x * blockDim.x + threadIdx.x;
    if (i < NNODES * DD) out[i] = agg[i] + b[i & (DD - 1)];
}

// Pack fp32 weight [256x256] row-major into bf16 WMMA B-fragment order:
//   Bp[((ct*8 + ks)*32 + lane)*16 + e] = W[ks*32 + (lane>>4)*16 + e][ct*16 + (lane&15)]
__global__ void k_pack_w(const float* __restrict__ W, bf16_t* __restrict__ Bp) {
    int idx = blockIdx.x * blockDim.x + threadIdx.x;   // 65536 total
    int e  = idx & 15;
    int l  = (idx >> 4) & 31;
    int ks = (idx >> 9) & 7;
    int ct = idx >> 12;
    int k   = ks * 32 + (l >> 4) * 16 + e;
    int col = ct * 16 + (l & 15);
    Bp[idx] = (bf16_t)W[k * DD + col];
}

// ---------------- WMMA GEMM: [M x 256] @ [256 x 256] ----------------
// Packed B staged in LDS (128 KB of the 320 KB/WGP pool); 8 waves / 128 rows per block.
// MODE 0: store f32 raw.  MODE 1: store bf16 with fused bias + relu.
template <int MODE>
__global__ __launch_bounds__(256) void k_gemm256(const bf16_t* __restrict__ A,
                                                 const bf16_t* __restrict__ Bp,
                                                 const float* __restrict__ bias,
                                                 float* __restrict__ outF,
                                                 bf16_t* __restrict__ outB, int M) {
    __shared__ bf16_t Bs[DD * DD];                  // 131072 bytes

    // cooperative stage: 8192 uint4 / 256 threads = 32 each
    {
        const uint4* src = (const uint4*)Bp;
        uint4* dst = (uint4*)Bs;
#pragma unroll
        for (int i = 0; i < 32; ++i)
            dst[threadIdx.x + i * 256] = src[threadIdx.x + i * 256];
    }
    __syncthreads();

    const int wave = threadIdx.x >> 5;
    const int lane = threadIdx.x & 31;
    const int rowBase = (blockIdx.x * 8 + wave) * 16;
    if (rowBase >= M) return;                       // wave-uniform, after barrier

    const int m  = lane & 15;
    const int kh = lane >> 4;
    int arow = rowBase + m;
    if (arow >= M) arow = M - 1;                    // clamped load, guarded store
    const bf16_t* ap = A + (size_t)arow * DD;

    v8f c[16];
#pragma unroll
    for (int t = 0; t < 16; ++t)
#pragma unroll
        for (int r = 0; r < 8; ++r) c[t][r] = 0.0f;

    // LDS fragment base for this lane: Bs[((ct*8 + ks)*32 + lane)*16]
    const bf16_t* bsl = Bs + (size_t)lane * 16;

    AFrag a, an;
    a.h[0] = *(const v8bf*)(ap + kh * 16);
    a.h[1] = *(const v8bf*)(ap + kh * 16 + 8);

    for (int ks = 0; ks < 8; ++ks) {
        if (ks < 7) {                               // prefetch next A fragment
            const int kb = (ks + 1) * 32 + kh * 16;
            an.h[0] = *(const v8bf*)(ap + kb);
            an.h[1] = *(const v8bf*)(ap + kb + 8);
        }
        const bf16_t* bp = bsl + (size_t)ks * 512;  // + ct*4096
        // 3-deep rotated pipeline over the 16 column tiles
        v16bf b0 = *(const v16bf*)(bp);
        v16bf b1 = *(const v16bf*)(bp + 4096);
#pragma unroll
        for (int ct = 0; ct < 16; ++ct) {
            v16bf b2 = b0;
            if (ct + 2 < 16) b2 = *(const v16bf*)(bp + (size_t)(ct + 2) * 4096);
            c[ct] = __builtin_amdgcn_wmma_f32_16x16x32_bf16(
                false, a.v, false, b0, (short)0, c[ct], false, false);
            b0 = b1;
            b1 = b2;
        }
        a = an;
    }

    const int col0 = lane & 15;
    const int mh   = (lane >> 4) * 8;
#pragma unroll
    for (int ct = 0; ct < 16; ++ct) {
#pragma unroll
        for (int r = 0; r < 8; ++r) {
            const int rr = rowBase + mh + r;
            if (rr < M) {
                const int col = ct * 16 + col0;
                float v = c[ct][r];
                if (MODE == 1) {
                    v += bias[col];
                    v = v > 0.0f ? v : 0.0f;
                    outB[(size_t)rr * DD + col] = (bf16_t)v;
                } else {
                    outF[(size_t)rr * DD + col] = v;
                }
            }
        }
    }
}

// ---------------- aggregation ----------------
__global__ void k_agg_init(const float* __restrict__ h, const float* __restrict__ dinv,
                           float* __restrict__ agg) {
    int i = blockIdx.x * blockDim.x + threadIdx.x;
    if (i < NNODES * DD) {
        float di = dinv[i >> 8];
        agg[i] = h[i] * di * di;                    // self loop term
    }
}
__global__ void k_agg_edges(const int* __restrict__ row, const int* __restrict__ col,
                            const float* __restrict__ dinv, const float* __restrict__ h,
                            float* __restrict__ agg) {
    size_t t = (size_t)blockIdx.x * blockDim.x + threadIdx.x;
    int e  = (int)(t >> 6);
    int c4 = (int)(t & 63);
    if (e >= NEDGES) return;
    int r = row[e], c = col[e];
    float nm = dinv[r] * dinv[c];
    const float4 v = *(const float4*)(h + (size_t)c * DD + c4 * 4);
    float* dst = agg + (size_t)r * DD + c4 * 4;
    atomicAdd(dst + 0, v.x * nm);
    atomicAdd(dst + 1, v.y * nm);
    atomicAdd(dst + 2, v.z * nm);
    atomicAdd(dst + 3, v.w * nm);
}

// ---------------- link predictor ----------------
__global__ void k_qgather(const int* __restrict__ e0, const int* __restrict__ e1,
                          const float* __restrict__ x, bf16_t* __restrict__ qb,
                          int q0, int qn) {
    int i = blockIdx.x * blockDim.x + threadIdx.x;
    if (i >= qn * DD) return;
    int q = i >> 8, d = i & (DD - 1);
    int a = e0[q0 + q], b = e1[q0 + q];
    qb[i] = (bf16_t)(x[(size_t)a * DD + d] * x[(size_t)b * DD + d]);
}
__global__ void k_final(const bf16_t* __restrict__ t2, const float* __restrict__ P2,
                        const float* __restrict__ pb2, float* __restrict__ out,
                        int q0, int qn) {
    int q = blockIdx.x * blockDim.x + threadIdx.x;
    if (q >= qn) return;
    const bf16_t* r = t2 + (size_t)q * DD;
    float s = pb2[0];
#pragma unroll 8
    for (int d = 0; d < DD; ++d) s += (float)r[d] * P2[d];
    out[q0 + q] = 1.0f / (1.0f + __expf(-s));
}

// ---------------- launcher ----------------
extern "C" void kernel_launch(void* const* d_in, const int* in_sizes, int n_in,
                              void* d_out, int out_size, void* d_ws, size_t ws_size,
                              hipStream_t stream) {
    const int*   adj_row = (const int*)d_in[0];
    const int*   adj_col = (const int*)d_in[1];
    const int*   edges   = (const int*)d_in[2];    // [2, Q] flat
    const float* emb     = (const float*)d_in[3];
    const float* W[3]    = {(const float*)d_in[4], (const float*)d_in[6], (const float*)d_in[8]};
    const float* b[3]    = {(const float*)d_in[5], (const float*)d_in[7], (const float*)d_in[9]};
    const float* P0  = (const float*)d_in[10];
    const float* pb0 = (const float*)d_in[11];
    const float* P1  = (const float*)d_in[12];
    const float* pb1 = (const float*)d_in[13];
    const float* P2  = (const float*)d_in[14];
    const float* pb2 = (const float*)d_in[15];
    float* out = (float*)d_out;

    // workspace carve-out (256B aligned)
    char* ws = (char*)d_ws;
    size_t off = 0;
    auto alloc = [&](size_t bytes) -> char* {
        off = (off + 255) & ~(size_t)255;
        char* p = ws + off;
        off += bytes;
        return p;
    };
    float*  dinv = (float*)alloc((size_t)NNODES * 4);
    bf16_t* xb   = (bf16_t*)alloc((size_t)NNODES * DD * 2);
    float*  h    = (float*)alloc((size_t)NNODES * DD * 4);
    float*  agg  = (float*)alloc((size_t)NNODES * DD * 4);
    bf16_t* wp[5];
    for (int i = 0; i < 5; ++i) wp[i] = (bf16_t*)alloc((size_t)DD * DD * 2);
    bf16_t* qb = (bf16_t*)alloc((size_t)QCH * DD * 2);
    bf16_t* t1 = (bf16_t*)alloc((size_t)QCH * DD * 2);
    bf16_t* t2 = (bf16_t*)alloc((size_t)QCH * DD * 2);
    (void)ws_size; (void)n_in; (void)in_sizes; (void)out_size;

    const int ND = NNODES * DD;

    // 1. degree + dinv
    k_deg_init<<<(NNODES + 255) / 256, 256, 0, stream>>>(dinv);
    k_deg_acc<<<(NEDGES + 255) / 256, 256, 0, stream>>>(adj_row, dinv);
    k_dinv<<<(NNODES + 255) / 256, 256, 0, stream>>>(dinv);

    // 2. bf16 conversions / weight packing
    k_f32_to_bf16<<<(ND + 255) / 256, 256, 0, stream>>>(emb, xb, ND);
    k_pack_w<<<256, 256, 0, stream>>>(W[0], wp[0]);
    k_pack_w<<<256, 256, 0, stream>>>(W[1], wp[1]);
    k_pack_w<<<256, 256, 0, stream>>>(W[2], wp[2]);
    k_pack_w<<<256, 256, 0, stream>>>(P0, wp[3]);
    k_pack_w<<<256, 256, 0, stream>>>(P1, wp[4]);

    // 3. three GCN layers
    const int gemmBlocksN = (NNODES + 127) / 128;
    const int edgeBlocks  = (int)(((size_t)NEDGES * 64 + 255) / 256);
    for (int L = 0; L < 3; ++L) {
        k_gemm256<0><<<gemmBlocksN, 256, 0, stream>>>(xb, wp[L], nullptr, h, nullptr, NNODES);
        k_agg_init<<<(ND + 255) / 256, 256, 0, stream>>>(h, dinv, agg);
        k_agg_edges<<<edgeBlocks, 256, 0, stream>>>(adj_row, adj_col, dinv, h, agg);
        if (L < 2)
            k_bias_relu_bf16<<<(ND + 255) / 256, 256, 0, stream>>>(agg, b[L], xb);
        else
            k_bias_f32<<<(ND + 255) / 256, 256, 0, stream>>>(agg, b[L], h);  // h := final node feats
    }

    // 4. link predictor in 8 query chunks
    const int gemmBlocksQ = (QCH + 127) / 128;
    for (int c = 0; c < 8; ++c) {
        int q0 = c * QCH;
        k_qgather<<<(QCH * DD + 255) / 256, 256, 0, stream>>>(edges, edges + NQ, h, qb, q0, QCH);
        k_gemm256<1><<<gemmBlocksQ, 256, 0, stream>>>(qb, wp[3], pb0, nullptr, t1, QCH);
        k_gemm256<1><<<gemmBlocksQ, 256, 0, stream>>>(t1, wp[4], pb1, nullptr, t2, QCH);
        k_final<<<(QCH + 255) / 256, 256, 0, stream>>>(t2, P2, pb2, out, q0, QCH);
    }
}
